// RetentionConv_5274219839543
// MI455X (gfx1250) — matrix operand order repaired and checked by reading, not compiled, not run on hardware
//
#include <hip/hip_runtime.h>

// ---------------------------------------------------------------------------
// RetentionConv for MI455X (gfx1250): all heavy math via v_wmma_f32_16x16x32_f16
// ---------------------------------------------------------------------------

typedef __attribute__((ext_vector_type(16))) _Float16 v16h;
typedef __attribute__((ext_vector_type(8)))  _Float16 v8h;
typedef __attribute__((ext_vector_type(8)))  float    v8f;

__device__ __forceinline__ v8f wmma32(v16h a, v16h b, v8f c) {
  // D = A(16x32 f16) * B(32x16 f16) + C(16x16 f32)
  return __builtin_amdgcn_wmma_f32_16x16x32_f16(
      /*neg_a=*/false, a, /*neg_b=*/false, b,
      /*c_mod=*/(short)0, c, /*reuse_a=*/false, /*reuse_b=*/false);
}

// A-operand fragment (16x32 f16), CDNA5 layout: each lane holds one row:
//   lanes 0-15 : row = row0+lane,    K = 0..7 and 16..23  (of the 32-chunk)
//   lanes 16-31: row = row0+lane-16, K = 8..15 and 24..31
// Two 16-byte loads per lane (global_load_b128 / ds_load_b128).
__device__ __forceinline__ v16h ldfragA(const _Float16* base, int ld,
                                        int row0, int k0, int lane) {
  int r = row0 + (lane & 15);
  const _Float16* p = base + (size_t)r * ld + k0 + ((lane >> 4) << 3);
  v8h lo = *(const v8h*)p;
  v8h hi = *(const v8h*)(p + 16);
  v16h out;
#pragma unroll
  for (int i = 0; i < 8; ++i) { out[i] = lo[i]; out[i + 8] = hi[i]; }
  return out;
}

// B-operand fragment (32x16 f16), CDNA5 layout: each lane holds one column;
// lanes 0-15 hold K=0..7 and 16..23, lanes 16-31 hold
// K=8..15 and 24..31 (same K order as ldfragA).  Lane holds column n = lane&15.
// `base` is indexed [n][k] with row stride `ld` halfs.
__device__ __forceinline__ v16h ldfragB(const _Float16* base, int ld,
                                        int n0, int k0, int lane) {
  int n = n0 + (lane & 15);
  const _Float16* p = base + (size_t)n * ld + k0 + ((lane >> 4) << 3);
  v8h lo = *(const v8h*)p;
  v8h hi = *(const v8h*)(p + 16);
  v16h out;
#pragma unroll
  for (int i = 0; i < 8; ++i) { out[i] = lo[i]; out[i + 8] = hi[i]; }
  return out;
}

// ------------------------------- f32 -> f16 --------------------------------
__global__ void cvt_f16(const float* __restrict__ src, _Float16* __restrict__ dst,
                        int n) {
  int i = blockIdx.x * blockDim.x + threadIdx.x;
  if (i < n) dst[i] = (_Float16)src[i];
}

// ------------------------- GEMM: C = A * W^T + bias ------------------------
// A: MxK f16 row-major.  W: NxK f16 row-major (so both operands contract along
// contiguous rows).  C: MxN f32.
// Block = 128 threads = 4 waves. Each wave computes a 16(M) x 64(N) strip,
// reusing its A fragment across 4 WMMA N-tiles per K-step.
__global__ void __launch_bounds__(128)
gemm_f16_nt(const _Float16* __restrict__ A, const _Float16* __restrict__ W,
            const float* __restrict__ bias, float* __restrict__ C,
            int M, int N, int K) {
  int wave = threadIdx.x >> 5;
  int lane = threadIdx.x & 31;
  int m0 = (blockIdx.x * 4 + wave) * 16;
  int n0 = blockIdx.y * 64;

  v8f acc[4] = {};
  for (int k0 = 0; k0 < K; k0 += 32) {
    v16h a = ldfragA(A, K, m0, k0, lane);
    if (k0 + 32 < K) {
      // hint the next A strip into cache (global_prefetch_b8)
      __builtin_prefetch(A + (size_t)(m0 + (lane & 15)) * K + k0 + 32, 0, 3);
    }
#pragma unroll
    for (int t = 0; t < 4; ++t) {
      v16h b = ldfragB(W, K, n0 + t * 16, k0, lane);
      acc[t] = wmma32(a, b, acc[t]);
    }
  }

  // C layout: VGPR i -> row m0+i (lanes 0-15) / m0+i+8 (lanes 16-31); col = lane&15
  int rh = (lane >> 4) * 8;
  int nl = lane & 15;
#pragma unroll
  for (int t = 0; t < 4; ++t) {
    int n = n0 + t * 16 + nl;
    float bv = bias[n];
#pragma unroll
    for (int i = 0; i < 8; ++i) {
      int m = m0 + i + rh;
      C[(size_t)m * N + n] = acc[t][i] + bv;
    }
  }
}

// --------------------- build attention operands (f16) ----------------------
// Aq[l] = [ q*cos(l*th) | q*sin(l*th) ]                      (L x 64 per b,h)
// Bk[m] = [ k*cos(m*th) | k*sin(m*th) ]                      (L x 64 per b,h)
// A2[l] = [ q*amp^{l+1}cos((l+1)th) | q*amp^{l+1}sin((l+1)th)]
// Vt[d][l] = v[l][d]   (transposed so B-fragments read contiguously along m)
__global__ void make_attn_inputs(const float* __restrict__ Q,
                                 const float* __restrict__ Kp,
                                 const float* __restrict__ V,
                                 const float* __restrict__ phazor,
                                 const float* __restrict__ amplitude,
                                 _Float16* __restrict__ Aq, _Float16* __restrict__ Bk,
                                 _Float16* __restrict__ A2, _Float16* __restrict__ Vt,
                                 int Bn, int H, int L, int D) {
  int idx = blockIdx.x * blockDim.x + threadIdx.x;
  int total = Bn * H * L * D;
  if (idx >= total) return;
  int d = idx & (D - 1);
  int l = (idx / D) & (L - 1);
  int h = (idx / (D * L)) % H;
  int b = idx / (D * L * H);

  float pr = phazor[2 * (h * D + d) + 0];
  float pi = phazor[2 * (h * D + d) + 1];
  float th = atan2f(pi, pr);                     // phase = e^{i*th}
  float amp = 1.f / (1.f + __expf(-amplitude[h]));

  float sl, cl, s1, c1;
  __sincosf((float)l * th, &sl, &cl);
  __sincosf((float)(l + 1) * th, &s1, &c1);
  float ampP = __powf(amp, (float)(l + 1));

  size_t col = (size_t)h * D + d;
  size_t row = (size_t)b * L + l;
  float q = Q[row * 512 + col];
  float k = Kp[row * 512 + col];
  float v = V[row * 512 + col];

  size_t base = ((size_t)(b * H + h) * L + l) * 64;
  Aq[base + d]      = (_Float16)(q * cl);
  Aq[base + 32 + d] = (_Float16)(q * sl);
  Bk[base + d]      = (_Float16)(k * cl);
  Bk[base + 32 + d] = (_Float16)(k * sl);
  A2[base + d]      = (_Float16)(q * ampP * c1);
  A2[base + 32 + d] = (_Float16)(q * ampP * s1);
  Vt[((size_t)(b * H + h) * D + d) * L + l] = (_Float16)v;
}

// B2t[h][e][k] : k<32 -> Re(last[h][k][e]) ; k>=32 -> -Im(last[h][k-32][e])
__global__ void make_B2(const float* __restrict__ last, _Float16* __restrict__ B2t,
                        int H) {
  int idx = blockIdx.x * blockDim.x + threadIdx.x;
  if (idx >= H * 32 * 64) return;
  int k = idx & 63;
  int e = (idx >> 6) & 31;
  int h = idx >> 11;
  float val;
  if (k < 32) val =  last[(((size_t)h * 32 + k) * 32 + e) * 2 + 0];
  else        val = -last[(((size_t)h * 32 + (k - 32)) * 32 + e) * 2 + 1];
  B2t[idx] = (_Float16)val;
}

// ------------------------------ attention ----------------------------------
// Per (b,h) and 16-row l-tile (one wave each; 4 waves per block):
//   for each 32-wide m-chunk with m <= l:
//     S = Aq_tile . Bk_chunk^T        (2 subtiles x 2 K-steps = 4 WMMA)
//     S *= amp^{l-m} causal mask      (f32, C layout)
//     stage S in LDS as f16, reload as A-fragment (DS ops in-order per wave)
//     inner += S . V_chunk            (2 WMMA over the two d-tiles)
//   inner += A2_tile . B2[h]          (4 WMMA, cross-chunk term)
//   write y tile to Yh f16 [(b*L+l)][h*32+d]
__global__ void __launch_bounds__(128)
attn_kernel(const _Float16* __restrict__ Aq, const _Float16* __restrict__ Bk,
            const _Float16* __restrict__ A2, const _Float16* __restrict__ Vt,
            const _Float16* __restrict__ B2t, const float* __restrict__ amplitude,
            _Float16* __restrict__ Yh, int Bn, int H, int L) {
  __shared__ __align__(16) _Float16 smem[4][16 * 40];  // 40-half row pitch
  int wave = threadIdx.x >> 5;
  int lane = threadIdx.x & 31;
  int bh = blockIdx.x;
  int h = bh & (H - 1);
  int b = bh / H;
  int lt = blockIdx.y * 4 + wave;
  int l0 = lt * 16;

  const _Float16* aq = Aq + (size_t)bh * L * 64;
  const _Float16* bk = Bk + (size_t)bh * L * 64;
  const _Float16* a2 = A2 + (size_t)bh * L * 64;
  const _Float16* vt = Vt + (size_t)bh * 32 * L;
  _Float16* sm = &smem[wave][0];

  float amp = 1.f / (1.f + __expf(-amplitude[h]));
  float logamp = __logf(amp);

  v16h aqf0 = ldfragA(aq, 64, l0, 0, lane);
  v16h aqf1 = ldfragA(aq, 64, l0, 32, lane);
  v8f acc0 = {};
  v8f acc1 = {};

  int rh = (lane >> 4) * 8;
  int nl = lane & 15;

  int nch = (l0 + 16 + 31) >> 5;  // only chunks intersecting m <= l
  for (int c = 0; c < nch; ++c) {
    int m0 = c * 32;
#pragma unroll
    for (int s = 0; s < 2; ++s) {
      v8f sc = {};
      // Bk as B operand: B[k][n] = Bk[m0+s*16+n][k], rows contiguous in k
      v16h b0 = ldfragB(bk, 64, m0 + s * 16, 0, lane);
      sc = wmma32(aqf0, b0, sc);
      v16h b1 = ldfragB(bk, 64, m0 + s * 16, 32, lane);
      sc = wmma32(aqf1, b1, sc);
      int m = m0 + s * 16 + nl;
#pragma unroll
      for (int i = 0; i < 8; ++i) {
        int l = l0 + i + rh;
        float val = (l >= m) ? sc[i] * __expf((float)(l - m) * logamp) : 0.f;
        sm[(i + rh) * 40 + s * 16 + nl] = (_Float16)val;
      }
    }
    // re-layout S (C-layout -> A-fragment) through LDS; per-wave slice,
    // DS instructions are executed in order within a wave.
    v16h sf = ldfragA(sm, 40, 0, 0, lane);
    // V as B operand: B[k][n] = Vt[d0+n][m0+k], contiguous along m
    v16h v0 = ldfragB(vt, L, 0, m0, lane);
    acc0 = wmma32(sf, v0, acc0);
    v16h v1 = ldfragB(vt, L, 16, m0, lane);
    acc1 = wmma32(sf, v1, acc1);
  }

  // cross-chunk term: A2_tile (16x64) x B2[h] (64x32)
  {
    v16h a2f0 = ldfragA(a2, 64, l0, 0, lane);
    v16h a2f1 = ldfragA(a2, 64, l0, 32, lane);
    const _Float16* b2 = B2t + (size_t)h * (32 * 64);
    acc0 = wmma32(a2f0, ldfragB(b2, 64, 0, 0, lane), acc0);
    acc0 = wmma32(a2f1, ldfragB(b2, 64, 0, 32, lane), acc0);
    acc1 = wmma32(a2f0, ldfragB(b2, 64, 16, 0, lane), acc1);
    acc1 = wmma32(a2f1, ldfragB(b2, 64, 16, 32, lane), acc1);
  }

  // y tile -> Yh[(b*L+l)][h*32+d]  (f16 input to the output projection GEMM)
#pragma unroll
  for (int i = 0; i < 8; ++i) {
    int l = l0 + i + rh;
    size_t rowb = ((size_t)(b * L + l)) * 512 + (size_t)h * 32;
    Yh[rowb + nl]      = (_Float16)acc0[i];
    Yh[rowb + 16 + nl] = (_Float16)acc1[i];
  }
}

// ------------------------------- launcher ----------------------------------
extern "C" void kernel_launch(void* const* d_in, const int* in_sizes, int n_in,
                              void* d_out, int out_size, void* d_ws, size_t ws_size,
                              hipStream_t stream) {
  (void)in_sizes; (void)n_in; (void)out_size; (void)ws_size;
  const float* x         = (const float*)d_in[0];
  const float* phazor    = (const float*)d_in[1];   // complex64 interleaved
  const float* amplitude = (const float*)d_in[2];
  const float* last      = (const float*)d_in[3];   // complex64 interleaved
  const float* wq_w = (const float*)d_in[4];
  const float* wq_b = (const float*)d_in[5];
  const float* wk_w = (const float*)d_in[6];
  const float* wk_b = (const float*)d_in[7];
  const float* wv_w = (const float*)d_in[8];
  const float* wv_b = (const float*)d_in[9];
  const float* wout_w = (const float*)d_in[10];
  const float* wout_b = (const float*)d_in[11];
  float* out = (float*)d_out;

  constexpr int Bn = 2, L = 1024, DIM = 512, H = 16, D = 32, HD = 512;
  constexpr int M = Bn * L;  // 2048

  char* ws = (char*)d_ws;
  size_t off = 0;
  auto alloc = [&](size_t bytes) {
    size_t r = off;
    off += (bytes + 255) & ~(size_t)255;
    return r;
  };
  _Float16* Xh  = (_Float16*)(ws + alloc((size_t)M * DIM * 2));
  _Float16* Wqh = (_Float16*)(ws + alloc((size_t)HD * DIM * 2));
  _Float16* Wkh = (_Float16*)(ws + alloc((size_t)HD * DIM * 2));
  _Float16* Wvh = (_Float16*)(ws + alloc((size_t)HD * DIM * 2));
  _Float16* Woh = (_Float16*)(ws + alloc((size_t)DIM * HD * 2));
  float*    Qf  = (float*)(ws + alloc((size_t)M * HD * 4));
  float*    Kf  = (float*)(ws + alloc((size_t)M * HD * 4));
  float*    Vf  = (float*)(ws + alloc((size_t)M * HD * 4));
  _Float16* Aq  = (_Float16*)(ws + alloc((size_t)Bn * H * L * 64 * 2));
  _Float16* Bk  = (_Float16*)(ws + alloc((size_t)Bn * H * L * 64 * 2));
  _Float16* A2  = (_Float16*)(ws + alloc((size_t)Bn * H * L * 64 * 2));
  _Float16* Vt  = (_Float16*)(ws + alloc((size_t)Bn * H * D * L * 2));
  _Float16* B2t = (_Float16*)(ws + alloc((size_t)H * 32 * 64 * 2));
  _Float16* Yh  = (_Float16*)(ws + alloc((size_t)M * HD * 2));

  // f32 -> f16 conversions
  cvt_f16<<<(M * DIM + 255) / 256, 256, 0, stream>>>(x, Xh, M * DIM);
  cvt_f16<<<(HD * DIM + 255) / 256, 256, 0, stream>>>(wq_w, Wqh, HD * DIM);
  cvt_f16<<<(HD * DIM + 255) / 256, 256, 0, stream>>>(wk_w, Wkh, HD * DIM);
  cvt_f16<<<(HD * DIM + 255) / 256, 256, 0, stream>>>(wv_w, Wvh, HD * DIM);
  cvt_f16<<<(DIM * HD + 255) / 256, 256, 0, stream>>>(wout_w, Woh, DIM * HD);

  // Q/K/V projections (WMMA)
  dim3 g(M / 64, HD / 64);
  gemm_f16_nt<<<g, 128, 0, stream>>>(Xh, Wqh, wq_b, Qf, M, HD, DIM);
  gemm_f16_nt<<<g, 128, 0, stream>>>(Xh, Wkh, wk_b, Kf, M, HD, DIM);
  gemm_f16_nt<<<g, 128, 0, stream>>>(Xh, Wvh, wv_b, Vf, M, HD, DIM);

  // phase-modulated operands + transposed V + cross-term B matrix
  int nel = Bn * H * L * D;
  make_attn_inputs<<<(nel + 255) / 256, 256, 0, stream>>>(
      Qf, Kf, Vf, phazor, amplitude, Aq, Bk, A2, Vt, Bn, H, L, D);
  make_B2<<<(H * 32 * 64 + 255) / 256, 256, 0, stream>>>(last, B2t, H);

  // retention attention (WMMA)
  dim3 ga(Bn * H, L / 64);
  attn_kernel<<<ga, 128, 0, stream>>>(Aq, Bk, A2, Vt, B2t, amplitude, Yh, Bn, H, L);

  // output projection (WMMA) -> d_out f32
  dim3 go(M / 64, DIM / 64);
  gemm_f16_nt<<<go, 128, 0, stream>>>(Yh, Woh, wout_b, out, M, DIM, HD);
}
